// CrossBlock_59115929862231
// MI455X (gfx1250) — compile-verified, matrix operand order checked by reading, not run on hardware
//
#include <hip/hip_runtime.h>
#include <hip/hip_bf16.h>

// ---------------------------------------------------------------------------
// CrossBlock for MI455X (gfx1250): bf16 WMMA everywhere, fused head-softmax
// attention (softmax is over the HEADS axis -> elementwise in (n,m), so no
// [N,N] logit tensor ever hits HBM). Software-pipelined GEMM (double-buffered
// LDS + async global->LDS copies tracked by ASYNCcnt) and attention loop
// (batched operand loads, early V loads, global_prefetch for next step).
// ---------------------------------------------------------------------------

typedef unsigned short u16;
typedef unsigned int   u32;
typedef __attribute__((ext_vector_type(16))) __bf16 v16bf;
typedef __attribute__((ext_vector_type(8)))  float  v8f;
typedef __attribute__((ext_vector_type(4)))  int    v4i;

union BV { v16bf v; uint4 q[2]; u32 u[8]; u16 s[16]; };

#define BATCH 2
#define SEQ   2048
#define DIM   512
#define NHEAD 8
#define HEADD 64
#define HIDD  2048
#define QK_SCALE 0.125f
#define LN_EPS 1e-5f

// ---- gfx1250 async global->LDS copy (ASYNCcnt path), with safe fallback ----
#if defined(__has_builtin)
#  if __has_builtin(__builtin_amdgcn_global_load_async_to_lds_b128)
#    define HAVE_ASYNC_COPY 1
#  endif
#  if __has_builtin(__builtin_amdgcn_s_wait_asynccnt)
#    define HAVE_WAIT_ASYNC 1
#  endif
#endif
#ifndef HAVE_ASYNC_COPY
#define HAVE_ASYNC_COPY 0
#endif
#ifndef HAVE_WAIT_ASYNC
#define HAVE_WAIT_ASYNC 0
#endif

__device__ __forceinline__ void copy16_g2lds(const u16* g, u16* l) {
#if HAVE_ASYNC_COPY
    // builtin takes int4-vector pointers: (global src, lds dst, imm off, imm cpol)
    __builtin_amdgcn_global_load_async_to_lds_b128((v4i*)g, (v4i*)l, 0, 0);
#else
    *(uint4*)l = *(const uint4*)g;
#endif
}

__device__ __forceinline__ void async_copy_wait() {
#if HAVE_ASYNC_COPY
#  if HAVE_WAIT_ASYNC
    __builtin_amdgcn_s_wait_asynccnt(0);
#  else
    asm volatile("s_wait_asynccnt 0x0" ::: "memory");
#  endif
#endif
}

__device__ __forceinline__ u16 f2bf(float f) {
    u32 u = __float_as_uint(f);
    return (u16)((u + 0x7FFFu + ((u >> 16) & 1u)) >> 16);
}

__device__ __forceinline__ v8f wmma_bf16(const BV& a, const BV& b, v8f c) {
    return __builtin_amdgcn_wmma_f32_16x16x32_bf16(
        /*neg_a=*/false, a.v, /*neg_b=*/false, b.v,
        /*c_mod=*/(short)0, c, /*reuse_a=*/false, /*reuse_b=*/false);
}

// ---------------------------------------------------------------------------
// Weight convert + transpose: W[K][N] f32 -> Wt[N][K] bf16
// ---------------------------------------------------------------------------
__global__ void k_transpose_bf16(const float* __restrict__ W,
                                 u16* __restrict__ Wt, int K, int N) {
    int i = blockIdx.x * blockDim.x + threadIdx.x;
    if (i >= K * N) return;
    int k = i / N, n = i % N;
    Wt[(size_t)n * K + k] = f2bf(W[i]);
}

// ---------------------------------------------------------------------------
// LayerNorm (rows of 512) -> bf16. One wave32 per row, 16 elems/lane.
// ---------------------------------------------------------------------------
__global__ __launch_bounds__(256)
void k_layernorm_bf16(const float* __restrict__ X, const float* __restrict__ w,
                      const float* __restrict__ bb, u16* __restrict__ out, int rows) {
    int row  = blockIdx.x * (blockDim.x >> 5) + (threadIdx.x >> 5);
    int lane = threadIdx.x & 31;
    if (row >= rows) return;
    const float4* xr = (const float4*)(X + (size_t)row * DIM);
    float4 v[4];
    float s = 0.f, s2 = 0.f;
#pragma unroll
    for (int i = 0; i < 4; ++i) {
        v[i] = xr[i * 32 + lane];
        s  += v[i].x + v[i].y + v[i].z + v[i].w;
        s2 += v[i].x * v[i].x + v[i].y * v[i].y + v[i].z * v[i].z + v[i].w * v[i].w;
    }
#pragma unroll
    for (int off = 16; off > 0; off >>= 1) {
        s  += __shfl_xor(s,  off, 32);
        s2 += __shfl_xor(s2, off, 32);
    }
    float mu  = s * (1.0f / DIM);
    float var = s2 * (1.0f / DIM) - mu * mu;
    float inv = rsqrtf(var + LN_EPS);
    uint2* orow = (uint2*)(out + (size_t)row * DIM);
#pragma unroll
    for (int i = 0; i < 4; ++i) {
        float4 wv = ((const float4*)w)[i * 32 + lane];
        float4 bv = ((const float4*)bb)[i * 32 + lane];
        float g0 = (v[i].x - mu) * inv * wv.x + bv.x;
        float g1 = (v[i].y - mu) * inv * wv.y + bv.y;
        float g2 = (v[i].z - mu) * inv * wv.z + bv.z;
        float g3 = (v[i].w - mu) * inv * wv.w + bv.w;
        uint2 p;
        p.x = (u32)f2bf(g0) | ((u32)f2bf(g1) << 16);
        p.y = (u32)f2bf(g2) | ((u32)f2bf(g3) << 16);
        orow[i * 32 + lane] = p;
    }
}

// ---------------------------------------------------------------------------
// Generic bf16 WMMA GEMM:  C[M][N] = A[M][K] @ Bt[N][K]^T  (+ epilogues)
// Block tile 128x64, 8 waves (4x2), wave tile 32x32 = 2x2 WMMA tiles, BK=32.
// Double-buffered LDS; next-tile staging (async global->LDS when available)
// overlaps current-tile WMMAs.
// EP: 0 = bf16, 1 = GELU->bf16, 2 = QKV scatter, 3 = f32 + bias + residual.
// ---------------------------------------------------------------------------
#define BM 128
#define BN 64
#define BK 32

template <int EP>
__global__ __launch_bounds__(256)
void k_gemm_bf16(const u16* __restrict__ A, const u16* __restrict__ Bt,
                 const float* __restrict__ bias, const float* __restrict__ resid,
                 float qscale, u16* __restrict__ outB,
                 u16* __restrict__ outQ, u16* __restrict__ outK,
                 u16* __restrict__ outV, float* __restrict__ outF,
                 int M, int N, int K) {
    __shared__ u16 As[2][BM][BK];
    __shared__ u16 Bs[2][BN][BK];

    int tid = threadIdx.x, lane = tid & 31, wid = tid >> 5;
    int wm = wid >> 1, wn = wid & 1;
    int rowBase = blockIdx.y * BM, colBase = blockIdx.x * BN;
    int rsel = lane & 15, kh = (lane >> 4) * 8;

    // per-thread staging coordinates
    int ar0 = (tid)       >> 2, asg0 = (tid)       & 3;
    int ar1 = (tid + 256) >> 2, asg1 = (tid + 256) & 3;
    int br  = tid >> 2,  bsg = tid & 3;
    const u16* gA0 = A  + (size_t)(rowBase + ar0) * K + asg0 * 8;
    const u16* gA1 = A  + (size_t)(rowBase + ar1) * K + asg1 * 8;
    const u16* gB  = Bt + (size_t)(colBase + br)  * K + bsg * 8;

    v8f acc[2][2];
#pragma unroll
    for (int i = 0; i < 2; ++i)
#pragma unroll
        for (int j = 0; j < 2; ++j) acc[i][j] = {};

    // prologue: stage tile 0 into buffer 0
    copy16_g2lds(gA0, &As[0][ar0][asg0 * 8]);
    copy16_g2lds(gA1, &As[0][ar1][asg1 * 8]);
    copy16_g2lds(gB,  &Bs[0][br][bsg * 8]);
    async_copy_wait();
    __syncthreads();

    int nk = K / BK;
    for (int kt = 0; kt < nk; ++kt) {
        int pb = kt & 1;
        // issue staging of the NEXT k-tile first (overlaps with WMMAs below)
        if (kt + 1 < nk) {
            int k0 = (kt + 1) * BK;
            copy16_g2lds(gA0 + k0, &As[pb ^ 1][ar0][asg0 * 8]);
            copy16_g2lds(gA1 + k0, &As[pb ^ 1][ar1][asg1 * 8]);
            copy16_g2lds(gB  + k0, &Bs[pb ^ 1][br][bsg * 8]);
        }

        BV a[2], b[2];
#pragma unroll
        for (int i = 0; i < 2; ++i) {
            a[i].q[0] = *(const uint4*)&As[pb][wm * 32 + i * 16 + rsel][kh];
            a[i].q[1] = *(const uint4*)&As[pb][wm * 32 + i * 16 + rsel][16 + kh];
        }
#pragma unroll
        for (int j = 0; j < 2; ++j) {
            b[j].q[0] = *(const uint4*)&Bs[pb][wn * 32 + j * 16 + rsel][kh];
            b[j].q[1] = *(const uint4*)&Bs[pb][wn * 32 + j * 16 + rsel][16 + kh];
        }
#pragma unroll
        for (int i = 0; i < 2; ++i)
#pragma unroll
            for (int j = 0; j < 2; ++j)
                acc[i][j] = wmma_bf16(a[i], b[j], acc[i][j]);

        async_copy_wait();   // next tile fully in LDS
        __syncthreads();
    }

    int rb = (lane >> 4) * 8;
#pragma unroll
    for (int i = 0; i < 2; ++i) {
#pragma unroll
        for (int j = 0; j < 2; ++j) {
            int r0 = rowBase + wm * 32 + i * 16 + rb;
            int c  = colBase + wn * 32 + j * 16 + (lane & 15);
            float bcol = bias ? bias[c] : 0.f;
#pragma unroll
            for (int r = 0; r < 8; ++r) {
                float val = acc[i][j][r] + bcol;
                int row = r0 + r;
                if (EP == 0) {
                    outB[(size_t)row * N + c] = f2bf(val);
                } else if (EP == 1) {
                    float g = 0.5f * val * (1.0f + erff(val * 0.70710678f));
                    outB[(size_t)row * N + c] = f2bf(g);
                } else if (EP == 2) {
                    int sec = c >> 9, c2 = c & 511;
                    int h = c2 >> 6, d = c2 & 63;
                    int bi = row >> 11, nn = row & 2047;
                    size_t qk = (((size_t)(bi * NHEAD + h)) * SEQ + nn) * HEADD + d;
                    if (sec == 0)      outQ[qk] = f2bf(val * qscale);
                    else if (sec == 1) outK[qk] = f2bf(val);
                    else outV[(((size_t)(bi * NHEAD + h)) * HEADD + d) * SEQ + nn] = f2bf(val);
                } else { // EP == 3: f32 + residual
                    outF[(size_t)row * N + c] = val + resid[(size_t)row * N + c];
                }
            }
        }
    }
}

// ---------------------------------------------------------------------------
// Fused dual cross-attention. Softmax is over heads (axis=1) -> elementwise
// per (n,m). Workgroup = (batch b, 16 n-rows); wave h = head h. Per 32-wide
// m-step: batched K loads -> 8 QK^T WMMAs -> gamma combine -> LDS 8-head
// softmax (V loads issued early to overlap) -> P@V WMMAs accumulated over m.
// ---------------------------------------------------------------------------
__global__ __launch_bounds__(256)
void k_attn_fused(const u16* __restrict__ Qx, const u16* __restrict__ Kx,
                  const u16* __restrict__ Vtx,
                  const u16* __restrict__ Qy, const u16* __restrict__ Ky,
                  const u16* __restrict__ Vty,
                  const float* __restrict__ g1p, const float* __restrict__ g2p,
                  const float* __restrict__ xin, const float* __restrict__ yin,
                  float* __restrict__ xo, float* __restrict__ yo) {
    __shared__ float scx[NHEAD][16][32];
    __shared__ float scy[NHEAD][16][32];
    __shared__ float smx[16][32], srx[16][32];
    __shared__ float smy[16][32], sry[16][32];

    int b  = blockIdx.x >> 7;
    int n0 = (blockIdx.x & 127) << 4;
    int lane = threadIdx.x & 31, h = threadIdx.x >> 5;
    int cc = lane & 15, kh = (lane >> 4) * 8, rb = (lane >> 4) * 8;
    float g1 = g1p[0], g2 = g2p[0];

    size_t headQ = ((size_t)(b * NHEAD + h)) * SEQ;

    // Loop-invariant Q A-operands (16 rows x 64 d), both streams.
    BV aqx[2], aqy[2];
    {
        const u16* qx = Qx + (headQ + n0 + cc) * HEADD;
        const u16* qy = Qy + (headQ + n0 + cc) * HEADD;
#pragma unroll
        for (int t = 0; t < 2; ++t) {
            aqx[t].q[0] = *(const uint4*)(qx + t * 32 + kh);
            aqx[t].q[1] = *(const uint4*)(qx + t * 32 + 16 + kh);
            aqy[t].q[0] = *(const uint4*)(qy + t * 32 + kh);
            aqy[t].q[1] = *(const uint4*)(qy + t * 32 + 16 + kh);
        }
    }

    // loop-carried base pointers -> all inner loads use immediate offsets
    const u16* pkx = Kx + (headQ + cc) * HEADD;
    const u16* pky = Ky + (headQ + cc) * HEADD;
    const u16* pvx = Vtx + (((size_t)(b * NHEAD + h)) * HEADD + cc) * SEQ;
    const u16* pvy = Vty + (((size_t)(b * NHEAD + h)) * HEADD + cc) * SEQ;

    v8f ox[4], oy[4];
#pragma unroll
    for (int t = 0; t < 4; ++t) { ox[t] = {}; oy[t] = {}; }

    for (int m0 = 0; m0 < SEQ; m0 += 32) {
        // prefetch next m-step tiles (global_prefetch_b8)
        __builtin_prefetch(pkx + 32 * HEADD, 0, 1);
        __builtin_prefetch(pky + 32 * HEADD, 0, 1);
        __builtin_prefetch(pvx + 32, 0, 1);
        __builtin_prefetch(pvy + 32, 0, 1);

        // ---- issue ALL K^T B-operand loads, then WMMA (staggered waits)
        BV bkx[2][2], bky[2][2];
#pragma unroll
        for (int mt = 0; mt < 2; ++mt)
#pragma unroll
            for (int kt = 0; kt < 2; ++kt) {
                const u16* px = pkx + mt * 16 * HEADD + kt * 32;
                const u16* py = pky + mt * 16 * HEADD + kt * 32;
                bkx[mt][kt].q[0] = *(const uint4*)(px + kh);
                bkx[mt][kt].q[1] = *(const uint4*)(px + 16 + kh);
                bky[mt][kt].q[0] = *(const uint4*)(py + kh);
                bky[mt][kt].q[1] = *(const uint4*)(py + 16 + kh);
            }
        v8f lx[2], ly[2], z = {};
#pragma unroll
        for (int mt = 0; mt < 2; ++mt) {
            lx[mt] = wmma_bf16(aqx[0], bkx[mt][0], z);
            lx[mt] = wmma_bf16(aqx[1], bkx[mt][1], lx[mt]);
            ly[mt] = wmma_bf16(aqy[0], bky[mt][0], z);
            ly[mt] = wmma_bf16(aqy[1], bky[mt][1], ly[mt]);
        }

        // ---- combine with gammas, stage to LDS in [h][n][m] layout
#pragma unroll
        for (int mt = 0; mt < 2; ++mt)
#pragma unroll
            for (int r = 0; r < 8; ++r) {
                float vx = lx[mt][r] + g1 * ly[mt][r];
                float vy = ly[mt][r] + g2 * lx[mt][r];
                scx[h][rb + r][mt * 16 + cc] = vx;
                scy[h][rb + r][mt * 16 + cc] = vy;
            }
        __syncthreads();

        // ---- issue V B-operand loads EARLY: they overlap the whole softmax
        BV bvx[4], bvy[4];
#pragma unroll
        for (int dt = 0; dt < 4; ++dt) {
            const u16* vx = pvx + dt * 16 * SEQ;
            const u16* vy = pvy + dt * 16 * SEQ;
            bvx[dt].q[0] = *(const uint4*)(vx + kh);
            bvx[dt].q[1] = *(const uint4*)(vx + 16 + kh);
            bvy[dt].q[0] = *(const uint4*)(vy + kh);
            bvy[dt].q[1] = *(const uint4*)(vy + 16 + kh);
        }

        // ---- per-position max & 1/sum over the 8 heads (cooperative)
#pragma unroll
        for (int p = 0; p < 2; ++p) {
            int flat = h * 64 + lane * 2 + p;
            int n = flat >> 5, m = flat & 31;
            float mxv = -3.0e38f, myv = -3.0e38f;
#pragma unroll
            for (int hh = 0; hh < NHEAD; ++hh) {
                mxv = fmaxf(mxv, scx[hh][n][m]);
                myv = fmaxf(myv, scy[hh][n][m]);
            }
            float sx = 0.f, sy = 0.f;
#pragma unroll
            for (int hh = 0; hh < NHEAD; ++hh) {
                sx += __expf(scx[hh][n][m] - mxv);
                sy += __expf(scy[hh][n][m] - myv);
            }
            smx[n][m] = mxv; srx[n][m] = 1.0f / sx;
            smy[n][m] = myv; sry[n][m] = 1.0f / sy;
        }
        __syncthreads();

        // ---- build P bf16 A-operands directly in WMMA A-layout
        BV apx, apy;
#pragma unroll
        for (int g = 0; g < 2; ++g) {
            int ms = g * 16 + kh;
#pragma unroll
            for (int jj = 0; jj < 4; ++jj) {
                int m = ms + jj * 2;
                float px0 = __expf(scx[h][cc][m]     - smx[cc][m])     * srx[cc][m];
                float px1 = __expf(scx[h][cc][m + 1] - smx[cc][m + 1]) * srx[cc][m + 1];
                float py0 = __expf(scy[h][cc][m]     - smy[cc][m])     * sry[cc][m];
                float py1 = __expf(scy[h][cc][m + 1] - smy[cc][m + 1]) * sry[cc][m + 1];
                apx.u[g * 4 + jj] = (u32)f2bf(px0) | ((u32)f2bf(px1) << 16);
                apy.u[g * 4 + jj] = (u32)f2bf(py0) | ((u32)f2bf(py1) << 16);
            }
        }

        // ---- P @ V (V operands already resident)
#pragma unroll
        for (int dt = 0; dt < 4; ++dt) {
            ox[dt] = wmma_bf16(apx, bvx[dt], ox[dt]);
            oy[dt] = wmma_bf16(apy, bvy[dt], oy[dt]);
        }
        __syncthreads(); // LDS reused next m-step

        pkx += 32 * HEADD;
        pky += 32 * HEADD;
        pvx += 32;
        pvy += 32;
    }

    // ---- epilogue: merge heads + residual, f32
#pragma unroll
    for (int dt = 0; dt < 4; ++dt)
#pragma unroll
        for (int r = 0; r < 8; ++r) {
            int n = n0 + rb + r;
            int d = dt * 16 + cc;
            size_t idx = ((size_t)b * SEQ + n) * DIM + h * HEADD + d;
            xo[idx] = ox[dt][r] + xin[idx];
            yo[idx] = oy[dt][r] + yin[idx];
        }
}

// ---------------------------------------------------------------------------
// Host-side orchestration
// ---------------------------------------------------------------------------
extern "C" void kernel_launch(void* const* d_in, const int* in_sizes, int n_in,
                              void* d_out, int out_size, void* d_ws, size_t ws_size,
                              hipStream_t stream) {
    const float* x      = (const float*)d_in[0];
    const float* y      = (const float*)d_in[1];
    const float* ln1_w  = (const float*)d_in[2];
    const float* ln1_b  = (const float*)d_in[3];
    const float* wx_qkv = (const float*)d_in[4];
    const float* bx_qkv = (const float*)d_in[5];
    const float* wy_qkv = (const float*)d_in[6];
    const float* by_qkv = (const float*)d_in[7];
    const float* gamma1 = (const float*)d_in[8];
    const float* gamma2 = (const float*)d_in[9];
    const float* ln2_w  = (const float*)d_in[10];
    const float* ln2_b  = (const float*)d_in[11];
    const float* w11    = (const float*)d_in[12];
    const float* b11    = (const float*)d_in[13];
    const float* w12    = (const float*)d_in[14];
    const float* b12    = (const float*)d_in[15];
    const float* w21    = (const float*)d_in[16];
    const float* b21    = (const float*)d_in[17];
    const float* w22    = (const float*)d_in[18];
    const float* b22    = (const float*)d_in[19];

    const int ROWS = BATCH * SEQ;            // 4096
    char* ws = (char*)d_ws;
    size_t off = 0;
    auto alloc = [&](size_t bytes) -> char* {
        char* p = ws + off;
        off = (off + bytes + 255) & ~(size_t)255;
        return p;
    };

    u16* wxT  = (u16*)alloc((size_t)3 * DIM * DIM * 2);        // [1536][512]
    u16* wyT  = (u16*)alloc((size_t)3 * DIM * DIM * 2);
    u16* w11T = (u16*)alloc((size_t)HIDD * DIM * 2);           // [2048][512]
    u16* w12T = (u16*)alloc((size_t)DIM * HIDD * 2);           // [512][2048]
    u16* w21T = (u16*)alloc((size_t)HIDD * DIM * 2);
    u16* w22T = (u16*)alloc((size_t)DIM * HIDD * 2);
    u16* xn   = (u16*)alloc((size_t)ROWS * DIM * 2);
    u16* yn   = (u16*)alloc((size_t)ROWS * DIM * 2);
    u16* Qx   = (u16*)alloc((size_t)ROWS * DIM * 2);
    u16* Kxb  = (u16*)alloc((size_t)ROWS * DIM * 2);
    u16* Vtx  = (u16*)alloc((size_t)ROWS * DIM * 2);
    u16* Qy   = (u16*)alloc((size_t)ROWS * DIM * 2);
    u16* Kyb  = (u16*)alloc((size_t)ROWS * DIM * 2);
    u16* Vty  = (u16*)alloc((size_t)ROWS * DIM * 2);
    float* xo = (float*)alloc((size_t)ROWS * DIM * 4);
    float* yo = (float*)alloc((size_t)ROWS * DIM * 4);
    u16* x2n  = (u16*)alloc((size_t)ROWS * DIM * 2);
    u16* y2n  = (u16*)alloc((size_t)ROWS * DIM * 2);
    u16* hx   = (u16*)alloc((size_t)ROWS * HIDD * 2);
    u16* hy   = (u16*)alloc((size_t)ROWS * HIDD * 2);

    float* outx = (float*)d_out;
    float* outy = outx + (size_t)ROWS * DIM;

    // 1) weight convert + transpose (bf16, [N][K])
    auto tgrid = [](int n) { return dim3((n + 255) / 256); };
    k_transpose_bf16<<<tgrid(DIM * 3 * DIM), 256, 0, stream>>>(wx_qkv, wxT, DIM, 3 * DIM);
    k_transpose_bf16<<<tgrid(DIM * 3 * DIM), 256, 0, stream>>>(wy_qkv, wyT, DIM, 3 * DIM);
    k_transpose_bf16<<<tgrid(DIM * HIDD),    256, 0, stream>>>(w11, w11T, DIM, HIDD);
    k_transpose_bf16<<<tgrid(HIDD * DIM),    256, 0, stream>>>(w12, w12T, HIDD, DIM);
    k_transpose_bf16<<<tgrid(DIM * HIDD),    256, 0, stream>>>(w21, w21T, DIM, HIDD);
    k_transpose_bf16<<<tgrid(HIDD * DIM),    256, 0, stream>>>(w22, w22T, HIDD, DIM);

    // 2) LN1 -> bf16
    k_layernorm_bf16<<<ROWS / 8, 256, 0, stream>>>(x, ln1_w, ln1_b, xn, ROWS);
    k_layernorm_bf16<<<ROWS / 8, 256, 0, stream>>>(y, ln1_w, ln1_b, yn, ROWS);

    // 3) QKV GEMMs with scatter epilogue (Q pre-scaled, V transposed)
    dim3 gq(3 * DIM / BN, ROWS / BM);
    k_gemm_bf16<2><<<gq, 256, 0, stream>>>(xn, wxT, bx_qkv, nullptr, QK_SCALE,
                                           nullptr, Qx, Kxb, Vtx, nullptr,
                                           ROWS, 3 * DIM, DIM);
    k_gemm_bf16<2><<<gq, 256, 0, stream>>>(yn, wyT, by_qkv, nullptr, QK_SCALE,
                                           nullptr, Qy, Kyb, Vty, nullptr,
                                           ROWS, 3 * DIM, DIM);

    // 4) fused dual attention + residual
    k_attn_fused<<<BATCH * (SEQ / 16), 256, 0, stream>>>(
        Qx, Kxb, Vtx, Qy, Kyb, Vty, gamma1, gamma2, x, y, xo, yo);

    // 5) LN2 -> bf16
    k_layernorm_bf16<<<ROWS / 8, 256, 0, stream>>>(xo, ln2_w, ln2_b, x2n, ROWS);
    k_layernorm_bf16<<<ROWS / 8, 256, 0, stream>>>(yo, ln2_w, ln2_b, y2n, ROWS);

    // 6) MLP: GEMM1 + exact GELU -> bf16, GEMM2 + bias + residual -> f32 out
    dim3 g1(HIDD / BN, ROWS / BM);
    dim3 g2(DIM / BN, ROWS / BM);
    k_gemm_bf16<1><<<g1, 256, 0, stream>>>(x2n, w11T, b11, nullptr, 1.f,
                                           hx, nullptr, nullptr, nullptr, nullptr,
                                           ROWS, HIDD, DIM);
    k_gemm_bf16<3><<<g2, 256, 0, stream>>>(hx, w12T, b12, xo, 1.f,
                                           nullptr, nullptr, nullptr, nullptr, outx,
                                           ROWS, DIM, HIDD);
    k_gemm_bf16<1><<<g1, 256, 0, stream>>>(y2n, w21T, b21, nullptr, 1.f,
                                           hy, nullptr, nullptr, nullptr, nullptr,
                                           ROWS, HIDD, DIM);
    k_gemm_bf16<3><<<g2, 256, 0, stream>>>(hy, w22T, b22, yo, 1.f,
                                           nullptr, nullptr, nullptr, nullptr, outy,
                                           ROWS, DIM, HIDD);
}
